// ProbableConv2d_69741678952871
// MI455X (gfx1250) — matrix operand order repaired and checked by reading, not compiled, not run on hardware
//
#include <hip/hip_runtime.h>
#include <cmath>

#define B_ 8
#define CI_ 128
#define CO_ 256
#define H_ 56
#define W_ 56
#define P_ 9
#define HW_ (H_*W_)          // 3136
#define NPIX_ (B_*HW_)       // 25088
#define KDIM_ (CI_*9)        // 1152 = K of the implicit GEMM (ci*9 + kh*3 + kw)
#define KSTEPS_ (KDIM_/32)   // 36 wmma k-steps
#define ROWPAD_ 1160         // 1152 + 8 bf16 pad -> bank stagger of 4

typedef __attribute__((ext_vector_type(16))) __bf16 v16bf;
typedef __attribute__((ext_vector_type(8)))  __bf16 v8bf;
typedef __attribute__((ext_vector_type(8)))  float  v8f;

union V16 { v16bf v; v8bf h[2]; };

// ---------------- fp32 -> bf16 convert (grid-stride) ----------------
__global__ __launch_bounds__(256)
void cvt_bf16(const float* __restrict__ in, __bf16* __restrict__ out, int n) {
    int i = blockIdx.x * blockDim.x + threadIdx.x;
    int stride = gridDim.x * blockDim.x;
    for (; i < n; i += stride) out[i] = (__bf16)in[i];
}

// ---------------- observer: 1x1 conv -> tanh -> 1x1 conv -> argmax ----------------
// one block = 32 consecutive pixels of one batch image (3136 % 32 == 0)
__global__ __launch_bounds__(256)
void observer_kernel(const float* __restrict__ x,
                     const float* __restrict__ ow1, const float* __restrict__ ob1,
                     const float* __restrict__ ow2, const float* __restrict__ ob2,
                     int* __restrict__ idx) {
    __shared__ float sX[CI_ * 32];
    __shared__ float sH[128 * 32];
    __shared__ float sS[P_ * 32];
    int tid = threadIdx.x;
    int pix0 = blockIdx.x * 32;
    int b = pix0 / HW_;
    int hw0 = pix0 - b * HW_;
    const float* xb = x + (size_t)b * CI_ * HW_ + hw0;

    for (int e = tid; e < CI_ * 32; e += 256) {      // coalesced 32-wide rows
        int c = e >> 5, pl = e & 31;
        sX[e] = xb[(size_t)c * HW_ + pl];
    }
    __syncthreads();

    int lane = tid & 31;          // pixel within tile
    int wv = tid >> 5;            // wave id (8 waves)
    for (int j = 0; j < 16; ++j) {
        int d = wv * 16 + j;
        float acc = ob1[d];
        const float* wrow = ow1 + d * CI_;           // uniform -> scalar loads
        for (int c = 0; c < CI_; ++c)
            acc = fmaf(wrow[c], sX[c * 32 + lane], acc);
        sH[d * 32 + lane] = tanhf(acc);
    }
    __syncthreads();

    if (tid < P_ * 32) {
        int p = tid >> 5, pl = tid & 31;
        float s = ob2[p];
        const float* wrow = ow2 + p * 128;
        for (int d = 0; d < 128; ++d)
            s = fmaf(wrow[d], sH[d * 32 + pl], s);
        sS[p * 32 + pl] = s;
    }
    __syncthreads();

    if (tid < 32) {
        float best = sS[tid];
        int bi = 0;
        for (int p = 1; p < P_; ++p) {               // strict '>' == jnp.argmax ties
            float s = sS[p * 32 + tid];
            if (s > best) { best = s; bi = p; }
        }
        idx[pix0 + tid] = bi;
    }
}

// ---------------- routing compaction ----------------
__global__ void zero_counts(int* cnt) { if (threadIdx.x < 16) cnt[threadIdx.x] = 0; }

__global__ __launch_bounds__(256)
void compact_kernel(const int* __restrict__ idx, int* __restrict__ cnt,
                    int* __restrict__ list) {
    int i = blockIdx.x * blockDim.x + threadIdx.x;
    if (i >= NPIX_) return;
    int p = idx[i];
    int pos = atomicAdd(&cnt[p], 1);   // order-independent: output deterministic
    list[p * NPIX_ + pos] = i;
}

// ---------------- grouped implicit-GEMM conv on the winning path ----------------
// grid = (pixel_tiles=784, P). Block = 256 threads = 8 waves.
// Tile: 32 pixels (same path, two N-tiles) x 256 CO, K = 1152.
// 2x2 register tiling: each wave owns 2 CO-tiles x 2 N-tiles; per k-step it
// loads A0,A1 (global) + B0,B1 (LDS) once and issues 4 wmma -> halves both
// L2 weight traffic and LDS reads per wmma vs a 1x1 tile.
__global__ __launch_bounds__(256)
void path_gemm_kernel(const __bf16* __restrict__ xb, const __bf16* __restrict__ wb,
                      const float* __restrict__ kb,
                      const int* __restrict__ cnt, const int* __restrict__ list,
                      float* __restrict__ out) {
    int p = blockIdx.y;
    int n0 = blockIdx.x * 32;
    int nv = cnt[p] - n0;        // scalar load; empty tiles exit immediately
    if (nv <= 0) return;
    if (nv > 32) nv = 32;

    __shared__ __align__(16) __bf16 sB[32 * ROWPAD_];   // [pixel][k] im2col patch
    __shared__ int sPix[32];

    int tid = threadIdx.x;
    if (tid < 32) sPix[tid] = (tid < nv) ? list[p * NPIX_ + n0 + tid] : 0;
    __syncthreads();

    // cooperative im2col gather: k = ci*9 + (kh*3+kw), matching kw memory order
    for (int e = tid; e < 32 * KDIM_; e += 256) {
        int n = e / KDIM_;
        int k = e - n * KDIM_;
        __bf16 val = (__bf16)0.f;
        if (n < nv) {
            int ci = k / 9;
            int pos = k - ci * 9;
            int kh = pos / 3, kw2 = pos - kh * 3;
            int pix = sPix[n];
            int b = pix / HW_;
            int hw = pix - b * HW_;
            int h = hw / W_, w = hw - h * W_;
            int hy = h + kh - 1, wx = w + kw2 - 1;
            if (hy >= 0 && hy < H_ && wx >= 0 && wx < W_)
                val = xb[((size_t)(b * CI_ + ci)) * HW_ + hy * W_ + wx];
        }
        sB[n * ROWPAD_ + k] = val;
    }
    __syncthreads();

    int lane = tid & 31;
    int wv = tid >> 5;
    int m = lane & 15;        // A row (co) AND B column (pixel) for this lane
    int hsel = lane >> 4;     // K-half selector per ISA VGPR layout

    int co0 = wv * 32;        // wave owns CO-tiles [co0, co0+16) and [co0+16, co0+32)
    v8f acc00 = {}, acc01 = {}, acc10 = {}, acc11 = {};
    const __bf16* arow0 = wb + ((size_t)(p * CO_ + co0 + m)) * KDIM_;
    const __bf16* arow1 = arow0 + (size_t)16 * KDIM_;
    const __bf16* brow0 = sB + m * ROWPAD_;
    const __bf16* brow1 = sB + (m + 16) * ROWPAD_;

    for (int kk = 0; kk < KSTEPS_; ++kk) {
        V16 a0, a1, b0, b1;
        // A 16x32 bf16: lane holds K = hsel*8+{0..7} and 16+hsel*8+{0..7}
        a0.h[0] = *(const v8bf*)(arow0 + kk * 32 + hsel * 8);
        a0.h[1] = *(const v8bf*)(arow0 + kk * 32 + 16 + hsel * 8);
        a1.h[0] = *(const v8bf*)(arow1 + kk * 32 + hsel * 8);
        a1.h[1] = *(const v8bf*)(arow1 + kk * 32 + 16 + hsel * 8);
        // B 32x16 bf16: lane holds K = hsel*8+{0..7} and 16+hsel*8+{0..7}
        b0.h[0] = *(const v8bf*)(brow0 + kk * 32 + hsel * 8);
        b0.h[1] = *(const v8bf*)(brow0 + kk * 32 + 16 + hsel * 8);
        b1.h[0] = *(const v8bf*)(brow1 + kk * 32 + hsel * 8);
        b1.h[1] = *(const v8bf*)(brow1 + kk * 32 + 16 + hsel * 8);
        acc00 = __builtin_amdgcn_wmma_f32_16x16x32_bf16(
            false, a0.v, false, b0.v, (short)0, acc00, false, false);
        acc01 = __builtin_amdgcn_wmma_f32_16x16x32_bf16(
            false, a0.v, false, b1.v, (short)0, acc01, false, false);
        acc10 = __builtin_amdgcn_wmma_f32_16x16x32_bf16(
            false, a1.v, false, b0.v, (short)0, acc10, false, false);
        acc11 = __builtin_amdgcn_wmma_f32_16x16x32_bf16(
            false, a1.v, false, b1.v, (short)0, acc11, false, false);
    }

    // C/D layout: lane column N = m (pixel), VGPR r -> row cobase + hsel*8 + r
    for (int tn = 0; tn < 2; ++tn) {
        int n = m + tn * 16;
        if (n >= nv) continue;
        int pix = sPix[n];
        int b = pix / HW_;
        int hw = pix - b * HW_;
        float* orow = out + (size_t)b * CO_ * HW_ + hw;
        for (int tc = 0; tc < 2; ++tc) {
            const v8f& acc = tc == 0 ? (tn == 0 ? acc00 : acc01)
                                     : (tn == 0 ? acc10 : acc11);
            int cobase = co0 + tc * 16;
            for (int r = 0; r < 8; ++r) {
                int co = cobase + hsel * 8 + r;
                float v = acc[r] + kb[p * CO_ + co];   // bias, then ReLU
                orow[(size_t)co * HW_] = v > 0.f ? v : 0.f;
            }
        }
    }
}

extern "C" void kernel_launch(void* const* d_in, const int* in_sizes, int n_in,
                              void* d_out, int out_size, void* d_ws, size_t ws_size,
                              hipStream_t stream) {
    const float* x   = (const float*)d_in[0];
    const float* kw  = (const float*)d_in[1];
    const float* kb  = (const float*)d_in[2];
    const float* ow1 = (const float*)d_in[3];
    const float* ob1 = (const float*)d_in[4];
    const float* ow2 = (const float*)d_in[5];
    const float* ob2 = (const float*)d_in[6];
    float* out = (float*)d_out;

    // workspace layout (all 64B-aligned offsets), ~12.1 MB total
    char* ws = (char*)d_ws;
    __bf16* xb  = (__bf16*)ws;                      // 3,211,264 bf16  (6.4 MB)
    __bf16* wbp = (__bf16*)(ws + 6422528);          // 2,654,208 bf16  (5.3 MB)
    int* idx    = (int*)(ws + 11730944);            // 25,088 int
    int* cnt    = (int*)(ws + 11831296);            // 16 int
    int* list   = (int*)(ws + 11831360);            // 9 * 25,088 int

    cvt_bf16<<<2048, 256, 0, stream>>>(x,  xb,  B_ * CI_ * HW_);
    cvt_bf16<<<2048, 256, 0, stream>>>(kw, wbp, P_ * CO_ * CI_ * 9);
    observer_kernel<<<NPIX_ / 32, 256, 0, stream>>>(x, ow1, ob1, ow2, ob2, idx);
    zero_counts<<<1, 32, 0, stream>>>(cnt);
    compact_kernel<<<(NPIX_ + 255) / 256, 256, 0, stream>>>(idx, cnt, list);

    dim3 grid((NPIX_ + 31) / 32, P_);
    path_gemm_kernel<<<grid, 256, 0, stream>>>(xb, wbp, kb, cnt, list, out);
}
